// InteractionDecoder_66348654788736
// MI455X (gfx1250) — compile-verified, matrix operand order checked by reading, not run on hardware
//
#include <hip/hip_runtime.h>
#include <stdint.h>

typedef __attribute__((ext_vector_type(16))) _Float16 v16h;
typedef __attribute__((ext_vector_type(8)))  _Float16 v8h;
typedef __attribute__((ext_vector_type(8)))  float    v8f;

#define N 1024
#define R 128
#define NBUCKETS 4096
#define CAP 65536
#define KTOP 128

// workspace layout (bytes)
#define EE_OFF    0                          // float[N*N]      4 MiB (skip-diag expanded sigmoid)
#define ERH_OFF   (N * N * 4)                // _Float16[N*R]   256 KiB
#define HIST_OFF  (ERH_OFF + N * R * 2)      // u32[NBUCKETS]
#define SEL_OFF   (HIST_OFF + NBUCKETS * 4)  // u32 threshold bucket
#define CNT_OFF   (SEL_OFF + 4)              // u32 candidate count
#define CANDV_OFF (CNT_OFF + 4 + 8)          // float[CAP]
#define CANDI_OFF (CANDV_OFF + CAP * 4)      // u32[CAP] packed (r<<20)|(i<<10)|j

__device__ __forceinline__ float sigmoidf(float x) {
    return 1.0f / (1.0f + __expf(-x));
}

// CDNA5 async global->LDS copy, 16 bytes per enabled lane. Tracked by ASYNCcnt.
__device__ __forceinline__ void async_copy_b128(uint32_t lds_off, const void* gaddr) {
    asm volatile("global_load_async_to_lds_b128 %0, %1, off"
                 :: "v"(lds_off), "v"(gaddr) : "memory");
}
__device__ __forceinline__ void async_wait0() {
    asm volatile("s_wait_asynccnt 0" ::: "memory");
}
// LDS byte offset of a __shared__ object: flat->LDS mapping uses addr[31:0] (wave-relative).
__device__ __forceinline__ uint32_t lds_off_of(const void* p) {
    return (uint32_t)(uintptr_t)p;
}

// Expand skip-diagonal ee into full [N,N] sigmoid table; er -> sigmoid f16.
__global__ void id_prep(const float* __restrict__ ee, const float* __restrict__ er,
                        float* __restrict__ ee_pair, _Float16* __restrict__ erh) {
    int idx = blockIdx.x * 256 + threadIdx.x;
    if (idx < N * N) {
        int i = idx >> 10, j = idx & (N - 1);
        int e = j - ((i < j) ? 1 : 0);
        e = e < 0 ? 0 : (e > N - 2 ? N - 2 : e);
        ee_pair[idx] = sigmoidf(ee[i * (N - 1) + e]);
    }
    int k = idx - N * N;
    if (k >= 0 && k < N * R) {
        erh[k] = (_Float16)sigmoidf(er[k]);
    }
}

// Process one 16x16 accumulator tile of outer products.
// score>0.5 implies every sigmoid factor >0.5 (participation mask redundant);
// only i!=j remains.  MODE 0: LDS histogram.  MODE 1: collect >= thrVal.
template <int MODE>
__device__ __forceinline__ void process_tile(
    const v8f& c, const float* ew, int r, int gi_base, int gj, int half,
    unsigned* shist, float thrVal, unsigned* gcnt,
    float* __restrict__ cval, unsigned* __restrict__ cidx) {
#pragma unroll
    for (int v = 0; v < 8; ++v) {
        const int gi = gi_base + v;            // gi_base already includes half*8
        float s = ew[v] * c[v];
        if (MODE == 0) {
            if (gi != gj && s > 0.5f) {
                unsigned bits = __float_as_uint(s);  // [0x3F000000,0x3F800000)
                unsigned bk = (bits - 0x3F000000u) >> 11;
                if (bk > NBUCKETS - 1) bk = NBUCKETS - 1;
                atomicAdd(&shist[bk], 1u);
            }
        } else {
            // bucket >= thr  <=>  s >= thrVal  (positive floats order as bits)
            if (gi != gj && s > 0.5f && s >= thrVal) {
                unsigned p = atomicAdd(gcnt, 1u);
                if (p < CAP) {
                    cval[p] = s;
                    cidx[p] = ((unsigned)r << 20) | ((unsigned)gi << 10) | (unsigned)gj;
                }
            }
        }
    }
}

// Main sweep: per 64x64 (i,j) supertile, loop relations; 16x16 outer products
// via v_wmma_f32_16x16x32_f16 with only the K=0 slice populated.
// Software-pipelined: WMMA for relation k issues before the accumulator of
// relation k-1 is consumed, so candidate-processing VALU fills the WMMA->VALU
// hazard slots (f16 WMMA needs 4 coexec slots) instead of v_nops.
template <int MODE>
__global__ __launch_bounds__(256) void id_score_pass(
    const float* __restrict__ ee_pair, const _Float16* __restrict__ erh,
    unsigned* __restrict__ ghist, const unsigned* __restrict__ sel,
    unsigned* __restrict__ gcnt, float* __restrict__ cval,
    unsigned* __restrict__ cidx) {
    __shared__ __attribute__((aligned(16))) float    eet[64 * 64];  // 16 KiB
    __shared__ __attribute__((aligned(16))) _Float16 erI[64 * R];   // 16 KiB
    __shared__ __attribute__((aligned(16))) _Float16 erJ[64 * R];   // 16 KiB
    __shared__ unsigned shist[NBUCKETS];                            // 16 KiB

    const int tid = threadIdx.x;
    const int ib = (blockIdx.x >> 4) * 64;
    const int jb = (blockIdx.x & 15) * 64;

    if (MODE == 0)
        for (int b = tid; b < NBUCKETS; b += 256) shist[b] = 0;

    // Stage er slabs (contiguous 16 KiB each) and the ee 64x64 tile via
    // async global->LDS b128 transfers (ASYNCcnt).
    {
        const uint32_t ldsI = lds_off_of(erI);
        const uint32_t ldsJ = lds_off_of(erJ);
        const uint32_t ldsE = lds_off_of(eet);
        const char* srcI = (const char*)(erh + (size_t)ib * R);
        const char* srcJ = (const char*)(erh + (size_t)jb * R);
        for (int c = tid; c < 1024; c += 256) {
            async_copy_b128(ldsI + c * 16, srcI + c * 16);
            async_copy_b128(ldsJ + c * 16, srcJ + c * 16);
            int row = c >> 4, seg = (c & 15) * 4;
            async_copy_b128(ldsE + (row * 64 + seg) * 4,
                            ee_pair + (size_t)(ib + row) * N + jb + seg);
        }
        async_wait0();
    }
    __syncthreads();

    float thrVal = 0.5f;
    if (MODE == 1) thrVal = __uint_as_float(0x3F000000u + (sel[0] << 11));

    const int wave = tid >> 5, lane = tid & 31;
    const int half = lane >> 4, lm = lane & 15;

    v16h a = {};   // A 16x32 f16: lanes 0-15 row M=lane, element 0 = K0; rest stays 0
    v16h b = {};   // B 32x16 f16: lanes 0-15 col N=lane, element 0 = K0; rest stays 0

    for (int t = wave; t < 16; t += 8) {   // 16 tiles of 16x16, 2 per wave
        const int i0 = (t >> 2) * 16, j0 = (t & 3) * 16;
        const int gj = jb + j0 + lm;
        const int gi_base = ib + i0 + (half << 3);
        // r-invariant ee row values for this lane's 8 accumulator slots
        float ew[8];
#pragma unroll
        for (int v = 0; v < 8; ++v)
            ew[v] = eet[(i0 + v + (half << 3)) * 64 + (j0 + lm)];

        const v8h* arow = (const v8h*)&erI[(i0 + lm) * R];
        const v8h* brow = (const v8h*)&erJ[(j0 + lm) * R];
        for (int r8 = 0; r8 < R / 8; ++r8) {
            // one 16B LDS read serves 8 relations; upper half-wave selects 0
            v8h av = arow[r8];
            v8h bv = brow[r8];
            if (half) { av = (v8h){}; bv = (v8h){}; }

            // pipelined: WMMA(k) issues before tile(k-1) is processed
            a[0] = av[0];
            b[0] = bv[0];
            v8f cp = {};
            cp = __builtin_amdgcn_wmma_f32_16x16x32_f16(
                     false, a, false, b, (short)0, cp, false, false);
#pragma unroll
            for (int k = 1; k < 8; ++k) {
                a[0] = av[k];
                b[0] = bv[k];
                v8f cn = {};
                cn = __builtin_amdgcn_wmma_f32_16x16x32_f16(
                         false, a, false, b, (short)0, cn, false, false);
                process_tile<MODE>(cp, ew, r8 * 8 + k - 1, gi_base, gj, half,
                                   shist, thrVal, gcnt, cval, cidx);
                cp = cn;
            }
            process_tile<MODE>(cp, ew, r8 * 8 + 7, gi_base, gj, half,
                               shist, thrVal, gcnt, cval, cidx);
        }
    }
    if (MODE == 0) {
        __syncthreads();
        for (int bkt = tid; bkt < NBUCKETS; bkt += 256)
            if (shist[bkt]) atomicAdd(&ghist[bkt], shist[bkt]);
    }
}

// Find largest bucket B such that count(bucket >= B) >= KTOP.
__global__ void id_select_thr(const unsigned* __restrict__ ghist, unsigned* __restrict__ sel) {
    if (threadIdx.x == 0) {
        unsigned cum = 0, thr = 0;
        for (int b = NBUCKETS - 1; b >= 0; --b) {
            cum += ghist[b];
            if (cum >= KTOP) { thr = (unsigned)b; break; }
        }
        sel[0] = thr;  // if total < KTOP, stays 0 -> collect everything
    }
}

// Exact top-128 over <=CAP candidates; (value desc, index asc) order for determinism.
__global__ __launch_bounds__(256) void id_final_topk(
    const float* __restrict__ cval, const unsigned* __restrict__ cidx,
    const unsigned* __restrict__ gcnt, const int* __restrict__ labels,
    float* __restrict__ out) {
    __shared__ float bv[256];
    __shared__ unsigned bi[256];
    __shared__ float pv;
    __shared__ unsigned pi;
    const int tid = threadIdx.x;
    unsigned count = *gcnt;
    if (count > CAP) count = CAP;
    if (tid == 0) { pv = __int_as_float(0x7F800000); pi = 0xFFFFFFFFu; }
    int fill = 0;
    for (int k = 0; k < KTOP; ++k) {
        __syncthreads();
        const float prevV = pv;
        const unsigned prevI = pi;
        float best = -1.0f;
        unsigned besti = 0xFFFFFFFFu;
        for (unsigned c = tid; c < count; c += 256) {
            float v = cval[c];
            unsigned id = cidx[c];
            bool after = (v < prevV) || (v == prevV && id > prevI);
            if (after && (v > best || (v == best && id < besti))) { best = v; besti = id; }
        }
        bv[tid] = best; bi[tid] = besti;
        __syncthreads();
        for (int s = 128; s > 0; s >>= 1) {
            if (tid < s) {
                float ov = bv[tid + s];
                unsigned oi = bi[tid + s];
                if (ov > bv[tid] || (ov == bv[tid] && oi < bi[tid])) { bv[tid] = ov; bi[tid] = oi; }
            }
            __syncthreads();
        }
        if (tid == 0) {
            float v = bv[0];
            unsigned id = bi[0];
            float val; int rel, sj, ob;
            if (v > 0.0f) {
                val = v; rel = (int)(id >> 20); sj = (int)((id >> 10) & 1023); ob = (int)(id & 1023);
                pv = v; pi = id;
            } else {  // fewer than KTOP candidates: pad like top_k over zeros (flat order)
                unsigned flat = (unsigned)fill++;
                val = 0.0f;
                rel = (int)(flat / (N * N));
                unsigned rem = flat % (N * N);
                sj = (int)(rem / N);
                ob = (int)(rem % N);
            }
            out[k]       = val;
            out[128 + k] = (float)rel;
            out[256 + k] = (float)sj;
            out[384 + k] = (float)ob;
            out[512 + k] = (float)labels[sj];
            out[640 + k] = (float)labels[ob];
        }
    }
}

extern "C" void kernel_launch(void* const* d_in, const int* in_sizes, int n_in,
                              void* d_out, int out_size, void* d_ws, size_t ws_size,
                              hipStream_t stream) {
    const float* ee = (const float*)d_in[0];      // [1024, 1023]
    const float* er = (const float*)d_in[1];      // [1024, 128]
    const int* labels = (const int*)d_in[2];      // [1024]

    char* ws = (char*)d_ws;
    float*     ee_pair = (float*)(ws + EE_OFF);
    _Float16*  erh     = (_Float16*)(ws + ERH_OFF);
    unsigned*  ghist   = (unsigned*)(ws + HIST_OFF);
    unsigned*  sel     = (unsigned*)(ws + SEL_OFF);
    unsigned*  gcnt    = (unsigned*)(ws + CNT_OFF);
    float*     cval    = (float*)(ws + CANDV_OFF);
    unsigned*  cidx    = (unsigned*)(ws + CANDI_OFF);

    // zero hist + sel + count every call (graph replays do not re-poison)
    hipMemsetAsync(ghist, 0, NBUCKETS * 4 + 16, stream);

    id_prep<<<(N * N + N * R + 255) / 256, 256, 0, stream>>>(ee, er, ee_pair, erh);
    id_score_pass<0><<<256, 256, 0, stream>>>(ee_pair, erh, ghist, sel, gcnt, cval, cidx);
    id_select_thr<<<1, 32, 0, stream>>>(ghist, sel);
    id_score_pass<1><<<256, 256, 0, stream>>>(ee_pair, erh, ghist, sel, gcnt, cval, cidx);
    id_final_topk<<<1, 256, 0, stream>>>(cval, cidx, gcnt, labels, (float*)d_out);
}
